// TriangleAttentionEndingNode_40501541601304
// MI455X (gfx1250) — matrix operand order repaired and checked by reading, not compiled, not run on hardware
//
#include <hip/hip_runtime.h>
#include <hip/hip_bf16.h>

// TriangleAttentionEndingNode for MI455X (gfx1250), wave32 + WMMA f16.
// B=1, L=256, CZ=128, H=4, CH=32.

#define LL   256
#define CZC  128
#define NHH  4
#define CHC  32
#define HCC  128          // NHH*CHC
#define NTOK (LL*LL)      // 65536
#define LNEPS 1e-5f

typedef _Float16 v16h __attribute__((ext_vector_type(16)));
typedef _Float16 v8h  __attribute__((ext_vector_type(8)));
typedef _Float16 v4h  __attribute__((ext_vector_type(4)));
typedef float    v8f  __attribute__((ext_vector_type(8)));

// ---------------------------------------------------------------- WMMA helpers

__device__ __forceinline__ v8f wmma_f16(v16h a, v16h b, v8f c) {
  // D = A(16x32 f16) * B(32x16 f16) + C(16x16 f32)
  return __builtin_amdgcn_wmma_f32_16x16x32_f16(false, a, false, b,
                                                (short)0, c, false, false);
}

// A-fragment (16x32, f16, row-major source with leading dim `ld` halves).
// Layout: lanes 0-15 hold row M=lane, halves = K{0..15};
//         lanes 16-31 hold row M=lane-16, halves = K{16..31}.
__device__ __forceinline__ v16h load_frag_a(const _Float16* base, int ld) {
  const int lane = threadIdx.x & 31;
  const int row  = lane & 15;
  const int ko   = (lane >> 4) << 4;      // 0 or 16
  const _Float16* p = base + row * ld + ko;
  v8h lo = *(const v8h*)(p);
  v8h hi = *(const v8h*)(p + 8);
  return __builtin_shufflevector(lo, hi, 0,1,2,3,4,5,6,7,8,9,10,11,12,13,14,15);
}

// B-fragment (32x16) loaded from a B^T ([N][K], row-major, leading dim `ld`)
// source: lanes 0-15 hold col N=lane, K=0..15; lanes 16-31 col N=lane-16,
// K=16..31 (16 contiguous halves -> one 32B load).
__device__ __forceinline__ v16h load_frag_bt(const _Float16* bt, int ld) {
  const int lane = threadIdx.x & 31;
  const int col  = lane & 15;
  const int ko   = (lane >> 4) << 4;      // 0 or 16
  return *(const v16h*)(bt + col * ld + ko);
}

// ---------------------------------------------------------------- 1) LayerNorm

__global__ void k_layernorm(const float* __restrict__ z,
                            const float* __restrict__ g,
                            const float* __restrict__ b,
                            _Float16* __restrict__ zn) {
  const int wave = threadIdx.x >> 5, lane = threadIdx.x & 31;
  const int tok  = blockIdx.x * 8 + wave;            // one wave per token
  const float4 x = *(const float4*)(z + (size_t)tok * CZC + lane * 4);
  float s  = x.x + x.y + x.z + x.w;
  float s2 = x.x*x.x + x.y*x.y + x.z*x.z + x.w*x.w;
#pragma unroll
  for (int off = 16; off; off >>= 1) {
    s  += __shfl_xor(s,  off);
    s2 += __shfl_xor(s2, off);
  }
  const float mu   = s * (1.0f / CZC);
  const float var  = s2 * (1.0f / CZC) - mu * mu;
  const float rstd = rsqrtf(var + LNEPS);
  const float4 gg = *(const float4*)(g + lane * 4);
  const float4 bb = *(const float4*)(b + lane * 4);
  v4h o;
  o[0] = (_Float16)((x.x - mu) * rstd * gg.x + bb.x);
  o[1] = (_Float16)((x.y - mu) * rstd * gg.y + bb.y);
  o[2] = (_Float16)((x.z - mu) * rstd * gg.z + bb.z);
  o[3] = (_Float16)((x.w - mu) * rstd * gg.w + bb.w);
  *(v4h*)(zn + (size_t)tok * CZC + lane * 4) = o;
}

// ------------------------------------------- 2) weights -> f16, transposed [N][K]

__global__ void k_wcvt(const float* __restrict__ Wq, const float* __restrict__ Wk,
                       const float* __restrict__ Wv, const float* __restrict__ Wg,
                       const float* __restrict__ Wo, _Float16* __restrict__ wt) {
  const int id  = blockIdx.x * blockDim.x + threadIdx.x;   // 5 * 128 * 128
  const int mat = id >> 14, rem = id & 16383;
  const int n = rem >> 7, k = rem & 127;
  const float* W = (mat == 0) ? Wq : (mat == 1) ? Wk : (mat == 2) ? Wv
                 : (mat == 3) ? Wg : Wo;
  wt[(size_t)mat * 16384 + n * 128 + k] = (_Float16)W[k * 128 + n];
}

// --------------------------------------------------- 3) QKVG projection (WMMA)

__global__ void __launch_bounds__(256)
k_proj(const _Float16* __restrict__ zn, const _Float16* __restrict__ wt,
       const float* __restrict__ bg,
       _Float16* __restrict__ qd, _Float16* __restrict__ kd,
       _Float16* __restrict__ vtd, _Float16* __restrict__ gd) {
  const int mat  = blockIdx.y;                  // 0:Q 1:K 2:V 3:G
  const int wave = threadIdx.x >> 5, lane = threadIdx.x & 31;
  const int mt = wave & 3, nhh = wave >> 2;
  const int m0 = blockIdx.x * 64 + mt * 16;
  const int n0 = nhh * 64;
  const _Float16* A  = zn + (size_t)m0 * CZC;
  const _Float16* Bt = wt + (size_t)mat * 16384 + (size_t)n0 * CZC;

  v8f acc[4] = {{}, {}, {}, {}};
#pragma unroll
  for (int ks = 0; ks < 4; ++ks) {
    if (ks < 3) __builtin_prefetch(A + (ks + 1) * 32);   // global_prefetch_b8
    v16h a = load_frag_a(A + ks * 32, CZC);
#pragma unroll
    for (int nt = 0; nt < 4; ++nt) {
      v16h bf = load_frag_bt(Bt + (size_t)nt * 16 * CZC + ks * 32, CZC);
      acc[nt] = wmma_f16(a, bf, acc[nt]);
    }
  }

  const int lo = lane & 15, hi8 = (lane >> 4) * 8;
#pragma unroll
  for (int nt = 0; nt < 4; ++nt) {
#pragma unroll
    for (int r = 0; r < 8; ++r) {
      const int m = m0 + hi8 + r;            // token = i*L + j
      const int n = n0 + nt * 16 + lo;       // n = h*CH + c
      const int i = m >> 8, j = m & (LL - 1);
      const int h = n >> 5, c = n & 31;
      const float val = acc[nt][r];
      const size_t rowidx = ((size_t)(j * NHH + h) * LL + i) * CHC + c;
      if (mat == 0)      qd[rowidx] = (_Float16)val;
      else if (mat == 1) kd[rowidx] = (_Float16)val;
      else if (mat == 2) vtd[((size_t)(j * NHH + h) * CHC + c) * LL + i] = (_Float16)val;
      else               gd[rowidx] = (_Float16)(1.0f / (1.0f + __expf(-(val + bg[n]))));
    }
  }
}

// --------------------------------------------------------- 4) pair bias (GEMV)

__global__ void k_bias(const _Float16* __restrict__ zn, const float* __restrict__ Wb,
                       float* __restrict__ bias_f) {
  const int id = blockIdx.x * blockDim.x + threadIdx.x;   // NTOK * NHH
  const int t = id >> 2, h = id & 3;
  const int ktok = t >> 8, j = t & (LL - 1);
  const _Float16* row = zn + (size_t)t * CZC;
  float s = 0.f;
#pragma unroll 8
  for (int c = 0; c < CZC; ++c) s += (float)row[c] * Wb[c * NHH + h];
  bias_f[((size_t)(j * NHH + h)) * LL + ktok] = s;
}

// --------------------------------- 5) attention per (j,h): S=QK^T, softmax, PV

__global__ void __launch_bounds__(128)
k_attn(const _Float16* __restrict__ qg, const _Float16* __restrict__ kg,
       const _Float16* __restrict__ vt, const _Float16* __restrict__ gg,
       const float* __restrict__ bias_f, const int* __restrict__ pmask,
       _Float16* __restrict__ hid) {
  const int j = blockIdx.x, h = blockIdx.y;
  const int wave = threadIdx.x >> 5, lane = threadIdx.x & 31;
  const int lo = lane & 15, hi8 = (lane >> 4) * 8;
  const size_t head = (size_t)(j * NHH + h);
  const _Float16* Q    = qg + head * LL * CHC;   // [i][c]
  const _Float16* Kt   = kg + head * LL * CHC;   // [k][c]  (= B^T for S)
  const _Float16* Vt   = vt + head * CHC * LL;   // [c][k]  (= B^T for PV)
  const float*    bias = bias_f + head * LL;     // [k]

  __shared__ __align__(16) _Float16 sk[LL * CHC];      // K tile, 16 KB
  __shared__ __align__(32) _Float16 sp[4][16 * 264];   // normalized P per wave
  const float scale = 0.1767766952966369f;             // 1/sqrt(32)

  // ---- stage the whole K tile once per workgroup via async-to-LDS engine.
  // Generic pointers to LDS are {aperture_hi32, lds_offset32}; truncate for VDST.
  {
    const char* gsrc = (const char*)Kt + (size_t)threadIdx.x * 16;
    const unsigned ldst =
        (unsigned)(unsigned long long)(const void*)((const char*)sk + threadIdx.x * 16);
#pragma unroll
    for (int it = 0; it < 8; ++it) {   // 128 thr * 16 B * 8 = 16 KB
      const unsigned long long ga = (unsigned long long)(gsrc + it * 2048);
      const unsigned la = ldst + it * 2048;
      asm volatile("global_load_async_to_lds_b128 %0, %1, off"
                   :: "v"(la), "v"(ga) : "memory");
    }
    asm volatile("s_wait_asynccnt 0x0" ::: "memory");
  }
  __syncthreads();

  for (int t = 0; t < 4; ++t) {
    const int i0 = (wave * 4 + t) * 16;
    v16h aq = load_frag_a(Q + (size_t)i0 * CHC, CHC);

    // ---- S = Q K^T : 16 column tiles, one WMMA each (K-dim = 32 = CH)
    v8f s[16];
#pragma unroll
    for (int nt = 0; nt < 16; ++nt) {
      v16h bk = load_frag_bt(sk + nt * 16 * CHC, CHC);   // ds_load from LDS
      v8f zz = {};
      s[nt] = wmma_f16(aq, bk, zz);
    }

    // ---- scale + bias + key mask, then row softmax in registers
    float lmax[8], lsum[8];
#pragma unroll
    for (int r = 0; r < 8; ++r) { lmax[r] = -1e30f; lsum[r] = 0.f; }
#pragma unroll
    for (int nt = 0; nt < 16; ++nt) {
      const int kk = nt * 16 + lo;
      const float bv = bias[kk];
      const float mv = (pmask[kk * LL + j] != 0) ? 0.f : -1e9f;
#pragma unroll
      for (int r = 0; r < 8; ++r) {
        const float vv = s[nt][r] * scale + bv + mv;
        s[nt][r] = vv;
        lmax[r] = fmaxf(lmax[r], vv);
      }
    }
#pragma unroll
    for (int r = 0; r < 8; ++r) {
#pragma unroll
      for (int off = 8; off; off >>= 1)
        lmax[r] = fmaxf(lmax[r], __shfl_xor(lmax[r], off));
    }
#pragma unroll
    for (int nt = 0; nt < 16; ++nt) {
#pragma unroll
      for (int r = 0; r < 8; ++r) {
        const float e = __expf(s[nt][r] - lmax[r]);
        s[nt][r] = e;
        lsum[r] += e;
      }
    }
#pragma unroll
    for (int r = 0; r < 8; ++r) {
#pragma unroll
      for (int off = 8; off; off >>= 1) lsum[r] += __shfl_xor(lsum[r], off);
      lsum[r] = 1.0f / lsum[r];
    }

    // ---- bounce normalized P through LDS to regain A-fragment layout
#pragma unroll
    for (int nt = 0; nt < 16; ++nt) {
#pragma unroll
      for (int r = 0; r < 8; ++r)
        sp[wave][(hi8 + r) * 264 + nt * 16 + lo] = (_Float16)(s[nt][r] * lsum[r]);
    }
    __syncthreads();

    // ---- O = P V : 8 K-steps x 2 column tiles
    v8f o0 = {}, o1 = {};
#pragma unroll
    for (int kt = 0; kt < 8; ++kt) {
      v16h pa = load_frag_a(&sp[wave][kt * 32], 264);
      v16h v0 = load_frag_bt(Vt + kt * 32, LL);
      v16h v1 = load_frag_bt(Vt + (size_t)16 * LL + kt * 32, LL);
      o0 = wmma_f16(pa, v0, o0);
      o1 = wmma_f16(pa, v1, o1);
    }

    // ---- gate + scatter into hidden buffer [token][h*CH+c]
#pragma unroll
    for (int r = 0; r < 8; ++r) {
      const int ii = i0 + hi8 + r;
      const size_t gbase = (head * LL + ii) * CHC;
      const size_t obase = ((size_t)ii * LL + j) * HCC + h * CHC;
      const float g0 = (float)gg[gbase + lo];
      const float g1 = (float)gg[gbase + 16 + lo];
      hid[obase + lo]      = (_Float16)(o0[r] * g0);
      hid[obase + 16 + lo] = (_Float16)(o1[r] * g1);
    }
    __syncthreads();
  }
}

// ------------------------------------------------- 6) output projection (WMMA)

__global__ void __launch_bounds__(256)
k_out(const _Float16* __restrict__ hid, const _Float16* __restrict__ wt_o,
      const float* __restrict__ bo, const int* __restrict__ pmask,
      float* __restrict__ out) {
  const int wave = threadIdx.x >> 5, lane = threadIdx.x & 31;
  const int mt = wave & 3, nhh = wave >> 2;
  const int m0 = blockIdx.x * 64 + mt * 16;
  const int n0 = nhh * 64;
  const _Float16* A  = hid + (size_t)m0 * HCC;
  const _Float16* Bt = wt_o + (size_t)n0 * HCC;

  v8f acc[4] = {{}, {}, {}, {}};
#pragma unroll
  for (int ks = 0; ks < 4; ++ks) {
    if (ks < 3) __builtin_prefetch(A + (ks + 1) * 32);
    v16h a = load_frag_a(A + ks * 32, HCC);
#pragma unroll
    for (int nt = 0; nt < 4; ++nt) {
      v16h bf = load_frag_bt(Bt + (size_t)nt * 16 * HCC + ks * 32, HCC);
      acc[nt] = wmma_f16(a, bf, acc[nt]);
    }
  }

  const int lo = lane & 15, hi8 = (lane >> 4) * 8;
#pragma unroll
  for (int nt = 0; nt < 4; ++nt) {
#pragma unroll
    for (int r = 0; r < 8; ++r) {
      const int m = m0 + hi8 + r;
      const int n = n0 + nt * 16 + lo;
      float val = acc[nt][r] + bo[n];
      val *= (float)pmask[m];
      out[(size_t)m * HCC + n] = val;
    }
  }
}

// ---------------------------------------------------------------------- launch

extern "C" void kernel_launch(void* const* d_in, const int* in_sizes, int n_in,
                              void* d_out, int out_size, void* d_ws, size_t ws_size,
                              hipStream_t stream) {
  (void)in_sizes; (void)n_in; (void)out_size; (void)ws_size;
  const float* z   = (const float*)d_in[0];
  const int*   pm  = (const int*)d_in[1];
  const float* lng = (const float*)d_in[2];
  const float* lnb = (const float*)d_in[3];
  const float* Wq  = (const float*)d_in[4];
  const float* Wk  = (const float*)d_in[5];
  const float* Wv  = (const float*)d_in[6];
  const float* Wb  = (const float*)d_in[7];
  const float* Wg  = (const float*)d_in[8];
  const float* bg  = (const float*)d_in[9];
  const float* Wo  = (const float*)d_in[10];
  const float* bo  = (const float*)d_in[11];
  float* out = (float*)d_out;

  char* ws = (char*)d_ws;
  size_t off = 0;
  auto alloc = [&](size_t bytes) -> void* {
    void* p = ws + off;
    off += (bytes + 255) & ~(size_t)255;
    return p;
  };
  _Float16* zn_h   = (_Float16*)alloc((size_t)NTOK * CZC * 2);   // 16 MB
  _Float16* wt     = (_Float16*)alloc((size_t)5 * 128 * 128 * 2);
  _Float16* q_h    = (_Float16*)alloc((size_t)NTOK * HCC * 2);   // [j][h][i][c]
  _Float16* k_h    = (_Float16*)alloc((size_t)NTOK * HCC * 2);   // [j][h][k][c]
  _Float16* v_t    = (_Float16*)alloc((size_t)NTOK * HCC * 2);   // [j][h][c][k]
  _Float16* g_h    = (_Float16*)alloc((size_t)NTOK * HCC * 2);   // [j][h][i][c]
  float*    bias_f = (float*)  alloc((size_t)LL * NHH * LL * 4); // [j][h][k]
  _Float16* hid    = (_Float16*)alloc((size_t)NTOK * HCC * 2);   // [tok][hc]

  k_layernorm<<<NTOK / 8, 256, 0, stream>>>(z, lng, lnb, zn_h);
  k_wcvt<<<(5 * 16384) / 256, 256, 0, stream>>>(Wq, Wk, Wv, Wg, Wo, wt);
  k_proj<<<dim3(NTOK / 64, 4), 256, 0, stream>>>(zn_h, wt, bg, q_h, k_h, v_t, g_h);
  k_bias<<<(NTOK * NHH) / 256, 256, 0, stream>>>(zn_h, Wb, bias_f);
  k_attn<<<dim3(LL, NHH), 128, 0, stream>>>(q_h, k_h, v_t, g_h, bias_f, pm, hid);
  k_out<<<NTOK / 64, 256, 0, stream>>>(hid, wt + (size_t)4 * 16384, bo, pm, out);
}
